// QueryModel_10746008175001
// MI455X (gfx1250) — matrix operand order repaired and checked by reading, not compiled, not run on hardware
//
#include <hip/hip_runtime.h>
#include <math.h>
#include <stdint.h>

typedef long long i64;
typedef __attribute__((ext_vector_type(16))) _Float16 v16h;
typedef __attribute__((ext_vector_type(8)))  _Float16 v8h;
typedef __attribute__((ext_vector_type(8)))  float    v8f;
typedef __attribute__((ext_vector_type(4)))  float    v4f;

// ---------------- problem constants (from reference) ----------------
#define NP      200000   // predicates
#define NDJ     100000   // disjunctions
#define NNODE   50000    // nodes
#define NEDGE   200000   // edges
#define NB      2048     // batch graphs
#define GEPS    1e-7f

// ---------------- small utility kernels ----------------
__global__ void fill_kernel(float* __restrict__ p, float v, i64 n) {
    i64 i = (i64)blockIdx.x * blockDim.x + threadIdx.x;
    if (i < n) p[i] = v;
}

__device__ __forceinline__ void atomicMaxF(float* addr, float val) {
    int old = __float_as_int(*addr);
    while (__int_as_float(old) < val) {
        int assumed = old;
        old = atomicCAS((int*)addr, assumed, __float_as_int(val));
        if (old == assumed) break;
    }
}

// ---------------- weight packing into WMMA B-fragment layout ----------------
// Wp[((c*(N/16)+nb)*32 + lane)*16 + j] = f16(W[k, n]),
//   k = c*32 + (lane>>4)*16 + j,  n = nb*16 + (lane&15),  zero for k >= K.
__global__ void pack_w_kernel(const float* __restrict__ W, _Float16* __restrict__ Wp,
                              int K, int N, i64 total)
{
    i64 idx = (i64)blockIdx.x * blockDim.x + threadIdx.x;
    if (idx >= total) return;
    int j    = (int)(idx & 15);
    int lane = (int)((idx >> 4) & 31);
    i64 rest = idx >> 9;
    int nblk = N >> 4;
    int nb   = (int)(rest % nblk);
    int c    = (int)(rest / nblk);
    int kk   = c * 32 + ((lane >> 4) << 4) + j;
    int n    = (nb << 4) + (lane & 15);
    Wp[idx]  = (kk < K) ? (_Float16)W[(i64)kk * N + n] : (_Float16)0.f;
}

// ---------------- WMMA GEMM: out[M,N] = act(A[M,K] @ W[K,N] + bias[N]) ----------------
// 128 threads = 4 waves. Block tile 32 rows x 64 cols; each wave owns a 32x16 D tile
// (two v_wmma_f32_16x16x32_f16 per K-chunk, shared B fragment).
// A tile staged in f32 LDS. Full tiles: async global->LDS DMA, double buffered
// (GLOBAL_LOAD_ASYNC_TO_LDS_B128 + s_wait_asynccnt). Tails: guarded sync path.
// act: 0 = none, 1 = leaky_relu(0.01), 2 = relu
__global__ __launch_bounds__(128)
void wmma_gemm_kernel(const float* __restrict__ A, const _Float16* __restrict__ Wp,
                      const float* __restrict__ bias, float* __restrict__ out,
                      int M, int K, int N, int act)
{
    __shared__ float Asf[2][32][36];         // 144B row stride (16B aligned), 2 buffers
    const int tid   = threadIdx.x;
    const int lane  = tid & 31;
    const int wave  = tid >> 5;
    const int m0    = blockIdx.x * 32;
    const int ncol0 = blockIdx.y * 64 + wave * 16;
    const int nb    = ncol0 >> 4;
    const int nblk  = N >> 4;
    const int lrow  = lane & 15;
    const int lhi   = lane >> 4;
    const int K32   = (K + 31) >> 5;

    v8f c0 = {0.f, 0.f, 0.f, 0.f, 0.f, 0.f, 0.f, 0.f};
    v8f c1 = {0.f, 0.f, 0.f, 0.f, 0.f, 0.f, 0.f, 0.f};

    // per-wave async DMA of one 32x32 f32 chunk: 2 x (32 lanes x 16B) covering 8 rows
    auto issue_async = [&](int ch, int buf) {
#pragma unroll
        for (int it = 0; it < 2; ++it) {
            int row = (wave << 3) + (it << 2) + (lane >> 3);
            int qf  = (lane & 7) << 2;    // float index within row
            unsigned long long gp =
                (unsigned long long)(uintptr_t)&A[(i64)(m0 + row) * K + (ch << 5) + qf];
            unsigned lp = (unsigned)(uintptr_t)&Asf[buf][row][qf];
            asm volatile("global_load_async_to_lds_b128 %0, %1, off"
                         :: "v"(lp), "v"(gp) : "memory");
        }
    };
    // fragment build: read f32 tile rows from LDS, convert to f16
    auto make_frag = [&](int buf, int r) -> v16h {
        v4f x0 = *(const v4f*)&Asf[buf][r][lhi * 16];
        v4f x1 = *(const v4f*)&Asf[buf][r][lhi * 16 + 4];
        v4f x2 = *(const v4f*)&Asf[buf][r][lhi * 16 + 8];
        v4f x3 = *(const v4f*)&Asf[buf][r][lhi * 16 + 12];
        v16h a;
#pragma unroll
        for (int j = 0; j < 4; ++j) {
            a[j]      = (_Float16)x0[j];
            a[j + 4]  = (_Float16)x1[j];
            a[j + 8]  = (_Float16)x2[j];
            a[j + 12] = (_Float16)x3[j];
        }
        return a;
    };
    auto do_wmma = [&](int buf, int ch) {
        v16h b = *(const v16h*)&Wp[(((i64)ch * nblk + nb) * 32 + lane) * 16];
        v16h a0 = make_frag(buf, lrow);
        v16h a1 = make_frag(buf, 16 + lrow);
        c0 = __builtin_amdgcn_wmma_f32_16x16x32_f16(false, a0, false, b, (short)0, c0, false, false);
        c1 = __builtin_amdgcn_wmma_f32_16x16x32_f16(false, a1, false, b, (short)0, c1, false, false);
    };

    const bool rows_full = (m0 + 32 <= M);
    const bool use_async = rows_full && ((K & 31) == 0);

    if (use_async) {
        issue_async(0, 0);
        for (int ch = 0; ch < K32; ++ch) {
            if (ch + 1 < K32) {
                issue_async(ch + 1, (ch + 1) & 1);
                asm volatile("s_wait_asynccnt 0x2" ::: "memory");  // current chunk done
            } else {
                asm volatile("s_wait_asynccnt 0x0" ::: "memory");
            }
            __syncthreads();               // all waves' DMA portions visible
            do_wmma(ch & 1, ch);
            __syncthreads();               // protect buffer from next prefetch
        }
    } else {
        for (int ch = 0; ch < K32; ++ch) {
            const int k0 = ch << 5;
            for (int e = tid; e < 32 * 32; e += 128) {
                int r = e >> 5, kk = e & 31;
                int gm = m0 + r, gk = k0 + kk;
                Asf[0][r][kk] = (gm < M && gk < K) ? A[(i64)gm * K + gk] : 0.f;
            }
            __syncthreads();
            do_wmma(0, ch);
            __syncthreads();
        }
    }

    // D layout: VGPR r, lane l -> m = r + 8*(l>>4), n = l&15
    const int gn = ncol0 + lrow;
    const float bv = bias[gn];
#pragma unroll
    for (int r = 0; r < 8; ++r) {
        int m = lhi * 8 + r;
        {
            int gm = m0 + m;
            if (gm < M) {
                float v = c0[r] + bv;
                if (act == 1)      v = v > 0.f ? v : 0.01f * v;
                else if (act == 2) v = v > 0.f ? v : 0.f;
                out[(i64)gm * N + gn] = v;
            }
        }
        {
            int gm = m0 + 16 + m;
            if (gm < M) {
                float v = c1[r] + bv;
                if (act == 1)      v = v > 0.f ? v : 0.01f * v;
                else if (act == 2) v = v > 0.f ? v : 0.f;
                out[(i64)gm * N + gn] = v;
            }
        }
    }
}

// ---------------- attention-aggregation pipeline ----------------
// gate logits: g[row] = dot(v[row,:], gW) + gb  (one wave per row, coalesced)
__global__ __launch_bounds__(256)
void gate_kernel(const float* __restrict__ v, const float* __restrict__ gW,
                 const float* __restrict__ gb, float* __restrict__ g,
                 int M, int K)
{
    int row  = blockIdx.x * (blockDim.x >> 5) + (threadIdx.x >> 5);
    int lane = threadIdx.x & 31;
    if (row >= M) return;
    float acc = 0.f;
    for (int k2 = lane; k2 < K; k2 += 32)
        acc += v[(i64)row * K + k2] * gW[k2];
#pragma unroll
    for (int off = 16; off > 0; off >>= 1)
        acc += __shfl_down(acc, off, 32);
    if (lane == 0) g[row] = acc + gb[0];
}

__global__ void seg_max_kernel(const float* __restrict__ g, const int* __restrict__ seg,
                               float* __restrict__ mx, int M)
{
    int i = blockIdx.x * blockDim.x + threadIdx.x;
    if (i >= M) return;
    atomicMaxF(&mx[seg[i]], g[i]);
}

__global__ void seg_expsum_kernel(const float* __restrict__ g, const float* __restrict__ mx,
                                  const int* __restrict__ seg, float* __restrict__ e,
                                  float* __restrict__ ssum, int M)
{
    int i = blockIdx.x * blockDim.x + threadIdx.x;
    if (i >= M) return;
    int s = seg[i];
    float ev = __expf(g[i] - mx[s]);
    e[i] = ev;
    atomicAdd(&ssum[s], ev);
}

__global__ void attn_scatter_kernel(const float* __restrict__ val, const float* __restrict__ e,
                                    const float* __restrict__ ssum, const int* __restrict__ seg,
                                    float* __restrict__ out, i64 M, int F)
{
    i64 idx = (i64)blockIdx.x * blockDim.x + threadIdx.x;
    if (idx >= M * (i64)F) return;
    i64 i = idx / F;
    int f = (int)(idx - i * F);
    int s = seg[i];
    float alpha = e[i] / (ssum[s] + 1e-16f);
    atomicAdd(&out[(i64)s * F + f], alpha * val[idx]);
}

// ---------------- concat kernels ----------------
__global__ void concat_node_kernel(const float* __restrict__ x, const float* __restrict__ pred,
                                   float* __restrict__ out, int n)
{
    i64 idx = (i64)blockIdx.x * blockDim.x + threadIdx.x;
    if (idx >= (i64)n * 160) return;
    i64 i = idx / 160; int f = (int)(idx - i * 160);
    out[idx] = (f < 32) ? x[i * 32 + f] : pred[i * 128 + (f - 32)];
}

__global__ void concat_edge_kernel(const float* __restrict__ attr, const float* __restrict__ dirs,
                                   float* __restrict__ out, int n)
{
    i64 idx = (i64)blockIdx.x * blockDim.x + threadIdx.x;
    if (idx >= (i64)n * 18) return;
    i64 i = idx / 18; int f = (int)(idx - i * 18);
    out[idx] = (f < 16) ? attr[i * 16 + f] : dirs[i * 2 + (f - 16)];
}

__global__ void concat_q_kernel(const float* __restrict__ q0, const float* __restrict__ q1,
                                const float* __restrict__ nc, float* __restrict__ out, int n)
{
    i64 idx = (i64)blockIdx.x * blockDim.x + threadIdx.x;
    if (idx >= (i64)n * 513) return;
    i64 i = idx / 513; int f = (int)(idx - i * 513);
    out[idx] = (f < 256) ? q0[i * 256 + f]
             : (f < 512) ? q1[i * 256 + (f - 256)]
                         : nc[i];
}

// ---------------- GENConv kernels ----------------
__global__ void gen_msg_kernel(const float* __restrict__ node_vec, float* __restrict__ msg,
                               const int* __restrict__ ei, i64 E2)
{
    i64 idx = (i64)blockIdx.x * blockDim.x + threadIdx.x;
    if (idx >= E2 * 256) return;
    i64 j = idx >> 8; int f = (int)(idx & 255);
    int src = ei[j];
    float v = node_vec[(i64)src * 256 + f] + msg[idx];
    msg[idx] = (v > 0.f ? v : 0.f) + GEPS;
}

__device__ __forceinline__ int gen_dst(const int* ei, i64 j, int E) {
    return (j < (i64)E) ? ei[(i64)E + j] : ei[j - (i64)E];
}

__global__ void gen_max_kernel(const float* __restrict__ msg, const int* __restrict__ ei,
                               const float* __restrict__ tptr, float* __restrict__ mx,
                               i64 E2, int E)
{
    i64 idx = (i64)blockIdx.x * blockDim.x + threadIdx.x;
    if (idx >= E2 * 256) return;
    i64 j = idx >> 8; int f = (int)(idx & 255);
    int dst = gen_dst(ei, j, E);
    atomicMaxF(&mx[(i64)dst * 256 + f], msg[idx] * tptr[0]);
}

__global__ void gen_sum_kernel(const float* __restrict__ msg, const int* __restrict__ ei,
                               const float* __restrict__ tptr, const float* __restrict__ mx,
                               float* __restrict__ sm, i64 E2, int E)
{
    i64 idx = (i64)blockIdx.x * blockDim.x + threadIdx.x;
    if (idx >= E2 * 256) return;
    i64 j = idx >> 8; int f = (int)(idx & 255);
    int dst = gen_dst(ei, j, E);
    float ev = __expf(msg[idx] * tptr[0] - mx[(i64)dst * 256 + f]);
    atomicAdd(&sm[(i64)dst * 256 + f], ev);
}

__global__ void gen_aggr_kernel(const float* __restrict__ msg, const int* __restrict__ ei,
                                const float* __restrict__ tptr, const float* __restrict__ mx,
                                const float* __restrict__ sm, float* __restrict__ aggr,
                                i64 E2, int E)
{
    i64 idx = (i64)blockIdx.x * blockDim.x + threadIdx.x;
    if (idx >= E2 * 256) return;
    i64 j = idx >> 8; int f = (int)(idx & 255);
    int dst = gen_dst(ei, j, E);
    i64 o = (i64)dst * 256 + f;
    float ev = __expf(msg[idx] * tptr[0] - mx[o]);
    float alpha = ev / (sm[o] + 1e-16f);
    atomicAdd(&aggr[o], alpha * msg[idx]);
}

__global__ void add_kernel(float* __restrict__ acc, const float* __restrict__ b, i64 n) {
    i64 i = (i64)blockIdx.x * blockDim.x + threadIdx.x;
    if (i < n) acc[i] = acc[i] + b[i];
}

// ---------------- launch ----------------
extern "C" void kernel_launch(void* const* d_in, const int* in_sizes, int n_in,
                              void* d_out, int out_size, void* d_ws, size_t ws_size,
                              hipStream_t stream)
{
    (void)in_sizes; (void)n_in; (void)out_size; (void)ws_size;

    int k = 0;
    const float* pred_W   = (const float*)d_in[k++]; // [32,64]
    const float* pred_b   = (const float*)d_in[k++]; // [64]
    const float* pagg_gW  = (const float*)d_in[k++]; // [64,1]
    const float* pagg_gb  = (const float*)d_in[k++]; // [1]
    const float* pagg_vW  = (const float*)d_in[k++]; // [64,64]
    const float* pagg_vb  = (const float*)d_in[k++]; // [64]
    const float* disj_W   = (const float*)d_in[k++]; // [64,64]
    const float* disj_b   = (const float*)d_in[k++]; // [64]
    const float* dagg_gW  = (const float*)d_in[k++]; // [64,1]
    const float* dagg_gb  = (const float*)d_in[k++]; // [1]
    const float* dagg_vW  = (const float*)d_in[k++]; // [64,128]
    const float* dagg_vb  = (const float*)d_in[k++]; // [128]
    const float* node_W   = (const float*)d_in[k++]; // [160,256]
    const float* node_b   = (const float*)d_in[k++]; // [256]
    const float* edge_W   = (const float*)d_in[k++]; // [18,256]
    const float* edge_b   = (const float*)d_in[k++]; // [256]
    const float* gen_t    = (const float*)d_in[k++]; // scalar
    const float* gen_W1   = (const float*)d_in[k++]; // [256,512]
    const float* gen_b1   = (const float*)d_in[k++]; // [512]
    const float* gen_W2   = (const float*)d_in[k++]; // [512,256]
    const float* gen_b2   = (const float*)d_in[k++]; // [256]
    const float* n0_gW    = (const float*)d_in[k++]; // [256,1]
    const float* n0_gb    = (const float*)d_in[k++]; // [1]
    const float* n0_vW    = (const float*)d_in[k++]; // [256,256]
    const float* n0_vb    = (const float*)d_in[k++]; // [256]
    const float* n1_gW    = (const float*)d_in[k++]; // [256,1]
    const float* n1_gb    = (const float*)d_in[k++]; // [1]
    const float* n1_vW    = (const float*)d_in[k++]; // [256,256]
    const float* n1_vb    = (const float*)d_in[k++]; // [256]
    const float* fin_W1   = (const float*)d_in[k++]; // [513,256]
    const float* fin_b1   = (const float*)d_in[k++]; // [256]
    const float* fin_W2   = (const float*)d_in[k++]; // [256,64]
    const float* fin_b2   = (const float*)d_in[k++]; // [64]
    const float* pvec     = (const float*)d_in[k++]; // [P,32]
    const int*   disj_idx = (const int*)  d_in[k++]; // [P]
    const int*   conj_idx = (const int*)  d_in[k++]; // [D]
    const float* xin      = (const float*)d_in[k++]; // [N,32]
    const float* eattr    = (const float*)d_in[k++]; // [E,16]
    const float* edirs    = (const float*)d_in[k++]; // [E,2]
    const float* edirs_r  = (const float*)d_in[k++]; // [E,2]
    const int*   eidx     = (const int*)  d_in[k++]; // [2,E] flat
    const int*   batch    = (const int*)  d_in[k++]; // [N]
    const float* ncount   = (const float*)d_in[k++]; // [B,1]
    float* outp = (float*)d_out;                     // [B,64]

    char* wsp = (char*)d_ws;
    auto alloc = [&](i64 nfloats) -> float* {
        float* p = (float*)wsp;
        wsp += ((size_t)nfloats * sizeof(float) + 255) & ~(size_t)255;
        return p;
    };
    auto alloch = [&](i64 nhalf) -> _Float16* {
        _Float16* p = (_Float16*)wsp;
        wsp += ((size_t)nhalf * sizeof(_Float16) + 255) & ~(size_t)255;
        return p;
    };
    auto kpad = [](int K) { return (i64)((K + 31) & ~31); };

    float* pv      = alloc((i64)NP * 64);
    float* pval    = alloc((i64)NP * 64);
    float* pg      = alloc(NP);
    float* pe      = alloc(NP);
    float* dmax    = alloc(NDJ);
    float* dsum    = alloc(NDJ);
    float* dvraw   = alloc((i64)NDJ * 64);
    float* dv      = alloc((i64)NDJ * 64);
    float* dg      = alloc(NDJ);
    float* de      = alloc(NDJ);
    float* dval    = alloc((i64)NDJ * 128);
    float* nmax    = alloc(NNODE);
    float* nsum    = alloc(NNODE);
    float* npred   = alloc((i64)NNODE * 128);
    float* ncat    = alloc((i64)NNODE * 160);
    float* nvec    = alloc((i64)NNODE * 256);
    float* ecat    = alloc((i64)NEDGE * 18);
    float* evec    = alloc((i64)2 * NEDGE * 256);   // reused in place as msg
    float* g0      = alloc(NNODE);
    float* e0      = alloc(NNODE);
    float* val0    = alloc((i64)NNODE * 256);
    float* bmax0   = alloc(NB);
    float* bsum0   = alloc(NB);
    float* q0      = alloc((i64)NB * 256);
    float* genmax  = alloc((i64)NNODE * 256);
    float* gensum  = alloc((i64)NNODE * 256);
    float* aggr    = alloc((i64)NNODE * 256);
    float* h1      = alloc((i64)NNODE * 512);
    float* nvec2   = alloc((i64)NNODE * 256);
    float* g1      = alloc(NNODE);
    float* e1      = alloc(NNODE);
    float* val1    = alloc((i64)NNODE * 256);
    float* bmax1   = alloc(NB);
    float* bsum1   = alloc(NB);
    float* q1      = alloc((i64)NB * 256);
    float* qcat    = alloc((i64)NB * 513);
    float* qh      = alloc((i64)NB * 256);

    _Float16* wp_pred = alloch(kpad(32)  * 64);
    _Float16* wp_pagg = alloch(kpad(64)  * 64);
    _Float16* wp_disj = alloch(kpad(64)  * 64);
    _Float16* wp_dagg = alloch(kpad(64)  * 128);
    _Float16* wp_node = alloch(kpad(160) * 256);
    _Float16* wp_edge = alloch(kpad(18)  * 256);
    _Float16* wp_gw1  = alloch(kpad(256) * 512);
    _Float16* wp_gw2  = alloch(kpad(512) * 256);
    _Float16* wp_n0v  = alloch(kpad(256) * 256);
    _Float16* wp_n1v  = alloch(kpad(256) * 256);
    _Float16* wp_fin1 = alloch(kpad(513) * 256);
    _Float16* wp_fin2 = alloch(kpad(256) * 64);

    const int T = 256;
    auto blocks = [&](i64 n) { return (unsigned)((n + T - 1) / T); };
    auto fill = [&](float* p, float v, i64 n) {
        fill_kernel<<<blocks(n), T, 0, stream>>>(p, v, n);
    };
    auto pack = [&](const float* W, _Float16* Wp, int K, int N) {
        i64 total = kpad(K) * N;
        pack_w_kernel<<<blocks(total), T, 0, stream>>>(W, Wp, K, N, total);
    };
    auto gemm = [&](const float* A, const _Float16* Wp, const float* b, float* out,
                    int M, int K, int N, int act) {
        dim3 grid((M + 31) / 32, N / 64);
        wmma_gemm_kernel<<<grid, 128, 0, stream>>>(A, Wp, b, out, M, K, N, act);
    };
    auto gate = [&](const float* v, const float* gW, const float* gb, float* g, int M, int K) {
        gate_kernel<<<(M * 32 + T - 1) / T, T, 0, stream>>>(v, gW, gb, g, M, K);
    };
    auto attn = [&](const float* val, const float* g, float* e, float* mx, float* sm,
                    const int* seg, float* out, int M, int S, int F) {
        fill(mx, -3.0e38f, S);
        fill(sm, 0.f, S);
        fill(out, 0.f, (i64)S * F);
        seg_max_kernel<<<blocks(M), T, 0, stream>>>(g, seg, mx, M);
        seg_expsum_kernel<<<blocks(M), T, 0, stream>>>(g, mx, seg, e, sm, M);
        attn_scatter_kernel<<<blocks((i64)M * F), T, 0, stream>>>(val, e, sm, seg, out, M, F);
    };

    // ---- pack all weights once ----
    pack(pred_W, wp_pred, 32, 64);
    pack(pagg_vW, wp_pagg, 64, 64);
    pack(disj_W, wp_disj, 64, 64);
    pack(dagg_vW, wp_dagg, 64, 128);
    pack(node_W, wp_node, 160, 256);
    pack(edge_W, wp_edge, 18, 256);
    pack(gen_W1, wp_gw1, 256, 512);
    pack(gen_W2, wp_gw2, 512, 256);
    pack(n0_vW, wp_n0v, 256, 256);
    pack(n1_vW, wp_n1v, 256, 256);
    pack(fin_W1, wp_fin1, 513, 256);
    pack(fin_W2, wp_fin2, 256, 64);

    // ---- predicate MLP + attention over disjunctions ----
    gemm(pvec, wp_pred, pred_b, pv, NP, 32, 64, 1);                      // lrelu
    gate(pv, pagg_gW, pagg_gb, pg, NP, 64);
    gemm(pv, wp_pagg, pagg_vb, pval, NP, 64, 64, 0);
    attn(pval, pg, pe, dmax, dsum, disj_idx, dvraw, NP, NDJ, 64);        // -> [D,64]

    // ---- disjunction MLP + attention over conjunctions ----
    gemm(dvraw, wp_disj, disj_b, dv, NDJ, 64, 64, 1);
    gate(dv, dagg_gW, dagg_gb, dg, NDJ, 64);
    gemm(dv, wp_dagg, dagg_vb, dval, NDJ, 64, 128, 0);
    attn(dval, dg, de, nmax, nsum, conj_idx, npred, NDJ, NNODE, 128);    // -> [N,128]

    // ---- node embedding ----
    concat_node_kernel<<<blocks((i64)NNODE * 160), T, 0, stream>>>(xin, npred, ncat, NNODE);
    gemm(ncat, wp_node, node_b, nvec, NNODE, 160, 256, 1);               // [N,256]

    // ---- edge embeddings (forward, then reverse, into one [2E,256] buffer) ----
    concat_edge_kernel<<<blocks((i64)NEDGE * 18), T, 0, stream>>>(eattr, edirs, ecat, NEDGE);
    gemm(ecat, wp_edge, edge_b, evec, NEDGE, 18, 256, 1);
    concat_edge_kernel<<<blocks((i64)NEDGE * 18), T, 0, stream>>>(eattr, edirs_r, ecat, NEDGE);
    gemm(ecat, wp_edge, edge_b, evec + (i64)NEDGE * 256, NEDGE, 18, 256, 1);

    // ---- pre-conv graph readout q0 ----
    gate(nvec, n0_gW, n0_gb, g0, NNODE, 256);
    gemm(nvec, wp_n0v, n0_vb, val0, NNODE, 256, 256, 0);
    attn(val0, g0, e0, bmax0, bsum0, batch, q0, NNODE, NB, 256);

    // ---- GENConv: per-feature segment softmax over destinations ----
    const i64 E2 = (i64)2 * NEDGE;
    gen_msg_kernel<<<blocks(E2 * 256), T, 0, stream>>>(nvec, evec, eidx, E2);
    fill(genmax, -3.0e38f, (i64)NNODE * 256);
    fill(gensum, 0.f, (i64)NNODE * 256);
    fill(aggr, 0.f, (i64)NNODE * 256);
    gen_max_kernel<<<blocks(E2 * 256), T, 0, stream>>>(evec, eidx, gen_t, genmax, E2, NEDGE);
    gen_sum_kernel<<<blocks(E2 * 256), T, 0, stream>>>(evec, eidx, gen_t, genmax, gensum, E2, NEDGE);
    gen_aggr_kernel<<<blocks(E2 * 256), T, 0, stream>>>(evec, eidx, gen_t, genmax, gensum, aggr, E2, NEDGE);
    add_kernel<<<blocks((i64)NNODE * 256), T, 0, stream>>>(aggr, nvec, (i64)NNODE * 256);
    gemm(aggr, wp_gw1, gen_b1, h1, NNODE, 256, 512, 2);                  // relu
    gemm(h1, wp_gw2, gen_b2, nvec2, NNODE, 512, 256, 1);                 // lrelu

    // ---- post-conv graph readout q1 ----
    gate(nvec2, n1_gW, n1_gb, g1, NNODE, 256);
    gemm(nvec2, wp_n1v, n1_vb, val1, NNODE, 256, 256, 0);
    attn(val1, g1, e1, bmax1, bsum1, batch, q1, NNODE, NB, 256);

    // ---- final MLP ----
    concat_q_kernel<<<blocks((i64)NB * 513), T, 0, stream>>>(q0, q1, ncount, qcat, NB);
    gemm(qcat, wp_fin1, fin_b1, qh, NB, 513, 256, 1);
    gemm(qh, wp_fin2, fin_b2, outp, NB, 256, 64, 1);
}
